// ProductionTPA_13357348291140
// MI455X (gfx1250) — matrix operand order repaired and checked by reading, not compiled, not run on hardware
//
#include <hip/hip_runtime.h>
#include <math.h>
#include <stdint.h>

#define BB 8
#define TT 8192
#define DD 512
#define HH 4
#define HDD 128
#define PP 24
#define PPAD 32
#define TOPKK 12
#define NCH 8
#define CHUNK 1024

typedef _Float16 f16;
typedef __attribute__((ext_vector_type(16))) _Float16 v16h;
typedef __attribute__((ext_vector_type(8)))  _Float16 v8h;
typedef __attribute__((ext_vector_type(8)))  float    v8f;
typedef __attribute__((ext_vector_type(4)))  float    v4f;
typedef __attribute__((ext_vector_type(4)))  unsigned int u32x4;
typedef __attribute__((ext_vector_type(8)))  int      i32x8;
typedef __attribute__((ext_vector_type(4)))  int      i32x4;

__device__ __forceinline__ v8f vzero8() {
  v8f z;
#pragma unroll
  for (int i = 0; i < 8; ++i) z[i] = 0.f;
  return z;
}

__device__ __forceinline__ v8f wmma_f16(v16h a, v16h b, v8f c) {
  return __builtin_amdgcn_wmma_f32_16x16x32_f16(false, a, false, b, (short)0, c,
                                                false, false);
}

// A-matrix fragment 16x32 f16 (ISA 7.12.2): lane m=l&15, half=l>>4,
// VGPR0-3 hold K=half*8..+7, VGPR4-7 hold K=16+half*8..+7 -> two 16B loads.
__device__ __forceinline__ v16h load_a_frag(const f16* __restrict__ p, int ld) {
  int lane = threadIdx.x & 31;
  int m = lane & 15, half = lane >> 4;
  const f16* r = p + (size_t)m * ld + half * 8;
  v8h lo = *(const v8h*)(r);
  v8h hi = *(const v8h*)(r + 16);
  v16h o;
#pragma unroll
  for (int i = 0; i < 8; ++i) { o[i] = lo[i]; o[i + 8] = hi[i]; }
  return o;
}

// B-matrix fragment 32x16 f16: lane n=l&15, half=l>>4, VGPR0-3 hold
// K=half*8..+7, VGPR4-7 hold K=16+half*8..+7 for column n -> two 16B loads.
__device__ __forceinline__ v16h load_b_frag(const f16* __restrict__ p, int ld) {
  int lane = threadIdx.x & 31;
  int n = lane & 15, half = lane >> 4;
  const f16* r = p + (size_t)n * ld + half * 8;
  v8h lo = *(const v8h*)(r);
  v8h hi = *(const v8h*)(r + 16);
  v16h o;
#pragma unroll
  for (int i = 0; i < 8; ++i) { o[i] = lo[i]; o[i + 8] = hi[i]; }
  return o;
}

// ---- TDM: DMA a dense [512 rows x 32 halves] weight slice into LDS.
// LDS rows padded 64B data + 16B pad (stride 80B) via D# pad fields to avoid
// 4-way bank conflicts on the ds_load_b128 fragment reads.
// D# layout per CDNA5 ISA ch.8 (group0 128b, group1 256b; groups 2/3 zero = 2D).
__device__ __forceinline__ void tdm_load_w_slice(const f16* gsrc, unsigned int lds_off) {
  uint64_t ga = (uint64_t)(size_t)gsrc;
  u32x4 g0;
  g0[0] = 1u;                                       // count=1, user descriptor
  g0[1] = lds_off;                                  // lds_addr (bytes)
  g0[2] = (unsigned int)ga;                         // global_addr[31:0]
  g0[3] = (unsigned int)((ga >> 32) & 0x1FFFFFFu)   // global_addr[56:32]
          | (2u << 30);                             // type=2 ("image")
  i32x8 g1;
  g1[0] = (int)((1u << 16)      // data_size = 2 bytes
                | (1u << 20)    // pad_enable
                | (3u << 22)    // pad_interval: every 16 DWORDs (64B)
                | (3u << 25));  // pad_amount: 4 DWORDs (16B)
  g1[1] = (int)(512u << 16);                 // tensor_dim0[15:0] @bits63:48
  g1[2] = (int)(512u << 16);                 // tensor_dim0 hi=0 | tensor_dim1 lo
  g1[3] = (int)(32u << 16);                  // tensor_dim1 hi=0 | tile_dim0=32
  g1[4] = 512;                               // tile_dim1=512, tile_dim2=0
  g1[5] = 512;                               // tensor_dim0_stride lo
  g1[6] = 0;                                 // stride0 hi | tensor_dim1_stride lo
  g1[7] = 0;
  i32x4 z4;
#pragma unroll
  for (int i = 0; i < 4; ++i) z4[i] = 0;
#if defined(__clang_major__) && (__clang_major__ >= 23)
  i32x8 z8;
#pragma unroll
  for (int i = 0; i < 8; ++i) z8[i] = 0;
  __builtin_amdgcn_tensor_load_to_lds(g0, g1, z4, z4, z8, 0);
#else
  __builtin_amdgcn_tensor_load_to_lds(g0, g1, z4, z4, 0);
#endif
}

// ---------------- weight f32 -> f16 conversion ----------------
__global__ void cvt_kernel(const float* __restrict__ a, const float* __restrict__ b,
                           f16* __restrict__ oa, f16* __restrict__ ob) {
  int i = blockIdx.x * 256 + threadIdx.x;  // 262144 elems each
  oa[i] = (f16)a[i];
  ob[i] = (f16)b[i];
}

// ---------------- LayerNorm -> xn f16, one wave per row ----------------
__global__ __launch_bounds__(256) void ln_kernel(const float* __restrict__ x,
                                                 const float* __restrict__ g,
                                                 const float* __restrict__ bb,
                                                 f16* __restrict__ xn) {
  int wave = threadIdx.x >> 5, lane = threadIdx.x & 31;
  int row = blockIdx.x * 8 + wave;
  const float* xr = x + (size_t)row * DD + lane * 16;
  v4f v[4];
#pragma unroll
  for (int i = 0; i < 4; ++i) v[i] = *(const v4f*)(xr + i * 4);
  float s = 0.f, ss = 0.f;
#pragma unroll
  for (int i = 0; i < 4; ++i)
#pragma unroll
    for (int j = 0; j < 4; ++j) { float q = v[i][j]; s += q; ss += q * q; }
#pragma unroll
  for (int m = 16; m >= 1; m >>= 1) {
    s += __shfl_xor(s, m, 32);
    ss += __shfl_xor(ss, m, 32);
  }
  float mu = s * (1.f / DD);
  float var = ss * (1.f / DD) - mu * mu;
  float rstd = rsqrtf(var + 1e-5f);
  const float* gp = g + lane * 16;
  const float* bp = bb + lane * 16;
  v8h o0, o1;
#pragma unroll
  for (int i = 0; i < 4; ++i)
#pragma unroll
    for (int j = 0; j < 4; ++j) {
      int e = i * 4 + j;
      float r = (v[i][j] - mu) * rstd * gp[e] + bp[e];
      if (e < 8) o0[e] = (f16)r; else o1[e - 8] = (f16)r;
    }
  f16* dst = xn + (size_t)row * DD + lane * 16;
  *(v8h*)dst = o0;
  *(v8h*)(dst + 8) = o1;
}

// ---------------- Qp = proto @ Wq^T, l2norm per head, fold 1/TEMP ----------------
__global__ __launch_bounds__(256) void qh_prep_kernel(const float* __restrict__ proto,
                                                      const float* __restrict__ wq,
                                                      f16* __restrict__ qh) {
  __shared__ float qp[PP][DD];
  __shared__ float scl[PP * HH];
  int t = threadIdx.x;
#pragma unroll 1
  for (int u = 0; u < 48; ++u) {
    int idx = t * 48 + u;       // 24*512 outputs
    int p = idx >> 9, j = idx & 511;
    const float* pr = proto + p * DD;
    const float* wr = wq + j * DD;
    float acc = 0.f;
    for (int d = 0; d < DD; ++d) acc += pr[d] * wr[d];
    qp[p][j] = acc;
  }
  __syncthreads();
  if (t < PP * HH) {
    int p = t >> 2, h = t & 3;
    float ssum = 0.f;
    for (int d = 0; d < HDD; ++d) { float q = qp[p][h * HDD + d]; ssum += q * q; }
    scl[t] = (1.0f / 0.07f) / fmaxf(sqrtf(ssum), 1e-12f);
  }
  __syncthreads();
#pragma unroll 1
  for (int u = 0; u < 64; ++u) {
    int idx = t * 64 + u;       // H*32*128
    int h = idx >> 12;
    int rem = idx & 4095;
    int p = rem >> 7, d = rem & 127;
    f16 val = (f16)0.f;
    if (p < PP) val = (f16)(qp[p][h * HDD + d] * scl[p * HH + h]);
    qh[idx] = val;
  }
}

// ---------------- WMMA projection: 64 tokens x 512 outs per WG ----------------
// Weight k-slices staged into LDS by the Tensor Data Mover, double-buffered on
// TENSORcnt. mode 0: K = xn @ Wk^T (per-head L2 normalized, row-major store).
// mode 1: V = xn @ Wv^T (transposed store Vt[(b,h,d)][t]).
#define WROW 40  // padded LDS row stride in halves (80B): 32 data + 8 pad
__global__ __launch_bounds__(256) void proj_kernel(const f16* __restrict__ xn,
                                                   const f16* __restrict__ w16,
                                                   f16* __restrict__ outp, int mode) {
  extern __shared__ char smem[];                    // 81920 bytes dynamic
  f16* wbuf0 = (f16*)smem;                          // 512 * 80B
  f16* wbuf1 = (f16*)(smem + 512 * 80);             // 512 * 80B
  f16* tile  = (f16*)smem;                          // aliased after k-loop: [64][264]
  unsigned int lds0 = (unsigned int)(size_t)(void*)wbuf0;
  unsigned int lds1 = (unsigned int)(size_t)(void*)wbuf1;

  int w = threadIdx.x >> 5, lane = threadIdx.x & 31;
  int tok0 = blockIdx.x * 64;
  int mb = w & 3, nh = w >> 2;
  int half = lane >> 4, n = lane & 15;

  v8f acc[16];
#pragma unroll
  for (int i = 0; i < 16; ++i) acc[i] = vzero8();

  if (w == 0) tdm_load_w_slice(w16, lds0);          // prime buffer 0 (kk=0)

  const f16* arow = xn + (size_t)(tok0 + mb * 16) * DD;
#pragma unroll 1
  for (int kk = 0; kk < 16; ++kk) {
    const f16* cur = (kk & 1) ? wbuf1 : wbuf0;
    if (w == 0) {
      if (kk < 15) {
        tdm_load_w_slice(w16 + (kk + 1) * 32, (kk & 1) ? lds0 : lds1);
        __builtin_amdgcn_s_wait_tensorcnt(1);       // current slice complete
      } else {
        __builtin_amdgcn_s_wait_tensorcnt(0);
      }
    }
    __syncthreads();                                // publish TDM data to all waves
    v16h a = load_a_frag(arow + kk * 32, DD);
    __builtin_prefetch(arow + (kk + 1) * 32, 0, 0);
#pragma unroll
    for (int nt = 0; nt < 16; ++nt) {
      v16h bf = load_b_frag(cur + (size_t)(nh * 256 + nt * 16) * WROW, WROW);
      acc[nt] = wmma_f16(a, bf, acc[nt]);
    }
    __syncthreads();                                // reads done before next DMA
  }

  if (mode == 0) {  // per-head L2 norm: head columns live entirely in this wave
#pragma unroll
    for (int r = 0; r < 8; ++r) {
      float s0 = 0.f, s1 = 0.f;
#pragma unroll
      for (int nt = 0; nt < 8; ++nt) {
        s0 += acc[nt][r] * acc[nt][r];
        s1 += acc[nt + 8][r] * acc[nt + 8][r];
      }
#pragma unroll
      for (int m = 1; m <= 8; m <<= 1) {
        s0 += __shfl_xor(s0, m, 32);
        s1 += __shfl_xor(s1, m, 32);
      }
      float c0 = 1.0f / fmaxf(sqrtf(s0), 1e-12f);
      float c1 = 1.0f / fmaxf(sqrtf(s1), 1e-12f);
#pragma unroll
      for (int nt = 0; nt < 8; ++nt) { acc[nt][r] *= c0; acc[nt + 8][r] *= c1; }
    }
  }

  int b = tok0 >> 13;
  int tloc = tok0 & (TT - 1);
#pragma unroll 1
  for (int rnd = 0; rnd < 2; ++rnd) {
    if (nh == rnd) {
#pragma unroll
      for (int nt = 0; nt < 16; ++nt)
#pragma unroll
        for (int r = 0; r < 8; ++r)
          tile[(size_t)(mb * 16 + half * 8 + r) * 264 + nt * 16 + n] = (f16)acc[nt][r];
    }
    __syncthreads();
    if (mode == 0) {
      int row = threadIdx.x >> 2, seg = threadIdx.x & 3;
      f16* dst = outp + (size_t)(tok0 + row) * DD + rnd * 256 + seg * 64;
      const f16* src = tile + (size_t)row * 264 + seg * 64;
#pragma unroll
      for (int c = 0; c < 8; ++c) *(v8h*)(dst + c * 8) = *(const v8h*)(src + c * 8);
    } else {
      int dglob = rnd * 256 + threadIdx.x;
      int h = dglob >> 7, dh = dglob & 127;
      f16* dst = outp + ((size_t)((b * HH + h) * HDD + dh)) * TT + tloc;
      f16 tmp[64];
#pragma unroll
      for (int j = 0; j < 64; ++j) tmp[j] = tile[(size_t)j * 264 + threadIdx.x];
#pragma unroll
      for (int c = 0; c < 8; ++c) {
        v8h vv;
#pragma unroll
        for (int j = 0; j < 8; ++j) vv[j] = tmp[c * 8 + j];
        *(v8h*)(dst + c * 8) = vv;
      }
    }
    __syncthreads();
  }
}

// ---------------- flash attention partials: WG=(b,h,chunk), wave-private T blocks ----------------
__global__ __launch_bounds__(256) void attn_kernel(const f16* __restrict__ qh,
                                                   const f16* __restrict__ kn,
                                                   const f16* __restrict__ vt,
                                                   float* __restrict__ pacc,
                                                   float* __restrict__ pml) {
  __shared__ f16 pb[8][2][16][32];
  int w = threadIdx.x >> 5, lane = threadIdx.x & 31;
  int c = blockIdx.x & 7;
  int h = (blockIdx.x >> 3) & 3;
  int b = blockIdx.x >> 5;
  int half = lane >> 4, n = lane & 15;

  v8f o[2][8];
  float m_run[2][8], l_run[2][8];
#pragma unroll
  for (int s = 0; s < 2; ++s)
#pragma unroll
    for (int r = 0; r < 8; ++r) { m_run[s][r] = -1e30f; l_run[s][r] = 0.f; }
#pragma unroll
  for (int s = 0; s < 2; ++s)
#pragma unroll
    for (int dt = 0; dt < 8; ++dt) o[s][dt] = vzero8();

  const f16* kb = kn + (size_t)b * TT * DD + h * HDD;
  const f16* vb = vt + (size_t)(b * HH + h) * HDD * TT;
  const f16* qb = qh + (size_t)(h * 32) * HDD;

#pragma unroll 1
  for (int i = 0; i < 4; ++i) {
    int t0 = c * CHUNK + (i * 8 + w) * 32;
    v8f sc[2][2];
#pragma unroll
    for (int a = 0; a < 2; ++a)
#pragma unroll
      for (int s = 0; s < 2; ++s) sc[a][s] = vzero8();

#pragma unroll
    for (int tt2 = 0; tt2 < 2; ++tt2) {
#pragma unroll
      for (int kk = 0; kk < 4; ++kk) {
        v16h bk = load_b_frag(kb + (size_t)(t0 + tt2 * 16) * DD + kk * 32, DD);
        v16h a0 = load_a_frag(qb + kk * 32, HDD);
        v16h a1 = load_a_frag(qb + (size_t)16 * HDD + kk * 32, HDD);
        sc[tt2][0] = wmma_f16(a0, bk, sc[tt2][0]);
        sc[tt2][1] = wmma_f16(a1, bk, sc[tt2][1]);
      }
    }

    // online softmax over the 32 new columns
#pragma unroll
    for (int s = 0; s < 2; ++s) {
#pragma unroll
      for (int r = 0; r < 8; ++r) {
        float x0 = sc[0][s][r], x1 = sc[1][s][r];
        float mx = fmaxf(x0, x1);
#pragma unroll
        for (int m = 1; m <= 8; m <<= 1) mx = fmaxf(mx, __shfl_xor(mx, m, 32));
        float newm = fmaxf(m_run[s][r], mx);
        float p0 = __expf(x0 - newm);
        float p1 = __expf(x1 - newm);
        float rs = p0 + p1;
#pragma unroll
        for (int m = 1; m <= 8; m <<= 1) rs += __shfl_xor(rs, m, 32);
        float alpha = __expf(m_run[s][r] - newm);
        l_run[s][r] = l_run[s][r] * alpha + rs;
        m_run[s][r] = newm;
        pb[w][s][half * 8 + r][n] = (f16)p0;
        pb[w][s][half * 8 + r][16 + n] = (f16)p1;
#pragma unroll
        for (int dt = 0; dt < 8; ++dt) o[s][dt][r] *= alpha;
      }
    }
    __syncthreads();  // make per-wave LDS probs visible cross-lane (uniform trip count)

    v16h pa0 = load_a_frag(&pb[w][0][0][0], 32);
    v16h pa1 = load_a_frag(&pb[w][1][0][0], 32);
#pragma unroll
    for (int dt = 0; dt < 8; ++dt) {
      v16h bv = load_b_frag(vb + (size_t)(dt * 16) * TT + t0, TT);
      o[0][dt] = wmma_f16(pa0, bv, o[0][dt]);
      o[1][dt] = wmma_f16(pa1, bv, o[1][dt]);
    }
  }

  int sidx = c * 8 + w;
  int bh = b * HH + h;
  int base = (bh * 64 + sidx) * 32;
  if (n == 0) {
#pragma unroll
    for (int s = 0; s < 2; ++s)
#pragma unroll
      for (int r = 0; r < 8; ++r) {
        int row = s * 16 + half * 8 + r;
        pml[(base + row) * 2 + 0] = m_run[s][r];
        pml[(base + row) * 2 + 1] = l_run[s][r];
      }
  }
#pragma unroll
  for (int s = 0; s < 2; ++s)
#pragma unroll
    for (int dt = 0; dt < 8; ++dt)
#pragma unroll
      for (int r = 0; r < 8; ++r) {
        int row = s * 16 + half * 8 + r;
        pacc[(size_t)(base + row) * 128 + dt * 16 + n] = o[s][dt][r];
      }
}

// ---------------- merge the 64 partials per (b,h) ----------------
__global__ __launch_bounds__(256) void combine_kernel(const float* __restrict__ pacc,
                                                      const float* __restrict__ pml,
                                                      float* __restrict__ ptok) {
  int bh = blockIdx.x;
  int b = bh >> 2, h = bh & 3;
  int p = threadIdx.x >> 3, dg = threadIdx.x & 7;
  int base0 = bh * 64;
  float M = -1e30f;
  for (int s = 0; s < 64; ++s) M = fmaxf(M, pml[((base0 + s) * 32 + p) * 2]);
  float v[16];
#pragma unroll
  for (int j = 0; j < 16; ++j) v[j] = 0.f;
  float ls = 0.f;
  for (int s = 0; s < 64; ++s) {
    int rb = (base0 + s) * 32 + p;
    float wgt = __expf(pml[rb * 2] - M);
    ls += pml[rb * 2 + 1] * wgt;
    const float* pa = pacc + (size_t)rb * 128 + dg * 16;
#pragma unroll
    for (int j = 0; j < 16; ++j) v[j] += pa[j] * wgt;
  }
  float inv = ls > 0.f ? 1.0f / ls : 0.f;
  if (p < PP) {
    float* dst = ptok + (size_t)(b * PPAD + p) * DD + h * HDD + dg * 16;
#pragma unroll
    for (int j = 0; j < 16; ++j) dst[j] = v[j] * inv;
  }
}

// ---------------- scores, top-k, mean, SiLU MLP (tiny) ----------------
__global__ __launch_bounds__(256) void head_kernel(const float* __restrict__ ptok,
                                                   const float* __restrict__ w1,
                                                   const float* __restrict__ b1,
                                                   const float* __restrict__ w2,
                                                   const float* __restrict__ b2,
                                                   float* __restrict__ out) {
  __shared__ float zs[DD];
  __shared__ float h1[DD];
  __shared__ float scb[PP];
  __shared__ int sel[TOPKK];
  int b = blockIdx.x, t = threadIdx.x;
  const float* pt = ptok + (size_t)b * PPAD * DD;
  if (t < PP) {
    const float* row = pt + t * DD;
    float ssum = 0.f;
    for (int d = 0; d < DD; ++d) ssum += row[d] * row[d];
    scb[t] = ssum;
  }
  __syncthreads();
  if (t == 0) {
    float tmp[PP];
    for (int p = 0; p < PP; ++p) tmp[p] = scb[p];
    for (int k = 0; k < TOPKK; ++k) {
      int bi = 0;
      float bv = tmp[0];
      for (int p = 1; p < PP; ++p)
        if (tmp[p] > bv) { bv = tmp[p]; bi = p; }
      sel[k] = bi;
      tmp[bi] = -1e30f;
    }
  }
  __syncthreads();
#pragma unroll
  for (int u = 0; u < 2; ++u) {
    int d = t + u * 256;
    float acc = 0.f;
    for (int k = 0; k < TOPKK; ++k) acc += pt[sel[k] * DD + d];
    zs[d] = acc * (1.0f / TOPKK);
  }
  __syncthreads();
#pragma unroll
  for (int u = 0; u < 2; ++u) {
    int j = t + u * 256;
    float acc = b1[j];
    const float* wr = w1 + (size_t)j * DD;
    for (int d = 0; d < DD; ++d) acc += zs[d] * wr[d];
    h1[j] = acc / (1.f + __expf(-acc));
  }
  __syncthreads();
#pragma unroll
  for (int u = 0; u < 2; ++u) {
    int oo = t + u * 256;
    float acc = b2[oo];
    const float* wr = w2 + (size_t)oo * DD;
    for (int j = 0; j < DD; ++j) acc += h1[j] * wr[j];
    out[b * DD + oo] = acc;
  }
}

extern "C" void kernel_launch(void* const* d_in, const int* in_sizes, int n_in,
                              void* d_out, int out_size, void* d_ws, size_t ws_size,
                              hipStream_t stream) {
  const float* x     = (const float*)d_in[0];
  const float* proto = (const float*)d_in[1];
  const float* ln_g  = (const float*)d_in[2];
  const float* ln_b  = (const float*)d_in[3];
  const float* Wq    = (const float*)d_in[4];
  const float* Wk    = (const float*)d_in[5];
  const float* Wv    = (const float*)d_in[6];
  const float* W1    = (const float*)d_in[7];
  const float* b1    = (const float*)d_in[8];
  const float* W2    = (const float*)d_in[9];
  const float* b2    = (const float*)d_in[10];
  float* out = (float*)d_out;

  char* ws = (char*)d_ws;
  size_t off = 0;
  auto alloc = [&](size_t bytes) -> char* {
    char* p = ws + off;
    off = (off + bytes + 255) & ~(size_t)255;
    return p;
  };
  f16* xn    = (f16*)alloc((size_t)BB * TT * DD * 2);
  f16* kn    = (f16*)alloc((size_t)BB * TT * DD * 2);
  f16* vt    = (f16*)alloc((size_t)BB * TT * DD * 2);
  f16* wk16  = (f16*)alloc((size_t)DD * DD * 2);
  f16* wv16  = (f16*)alloc((size_t)DD * DD * 2);
  f16* qh    = (f16*)alloc((size_t)HH * PPAD * HDD * 2);
  float* pacc = (float*)alloc((size_t)BB * HH * 64 * 32 * 128 * 4);
  float* pml  = (float*)alloc((size_t)BB * HH * 64 * 32 * 2 * 4);
  float* ptok = (float*)alloc((size_t)BB * PPAD * DD * 4);

  const size_t proj_lds = 512 * 80 * 2;  // double-buffered TDM weight slices

  cvt_kernel<<<(DD * DD) / 256, 256, 0, stream>>>(Wk, Wv, wk16, wv16);
  ln_kernel<<<(BB * TT) / 8, 256, 0, stream>>>(x, ln_g, ln_b, xn);
  qh_prep_kernel<<<1, 256, 0, stream>>>(proto, Wq, qh);
  proj_kernel<<<(BB * TT) / 64, 256, proj_lds, stream>>>(xn, wk16, kn, 0);
  proj_kernel<<<(BB * TT) / 64, 256, proj_lds, stream>>>(xn, wv16, vt, 1);
  attn_kernel<<<BB * HH * NCH, 256, 0, stream>>>(qh, kn, vt, pacc, pml);
  combine_kernel<<<BB * HH, 256, 0, stream>>>(pacc, pml, ptok);
  head_kernel<<<BB, 256, 0, stream>>>(ptok, W1, b1, W2, b2, out);
}
